// QuantCrossAttentionWithCudaKernel_78056735638209
// MI455X (gfx1250) — 16-bit and 8-bit matrix operand order repaired and checked by reading, compiled, not run on hardware
//
#include <hip/hip_runtime.h>

typedef signed char i8;
typedef __attribute__((ext_vector_type(2)))  int       v2i;
typedef __attribute__((ext_vector_type(4)))  int       v4i;
typedef __attribute__((ext_vector_type(8)))  int       v8i;
typedef __attribute__((ext_vector_type(8)))  float     v8f;
typedef __attribute__((ext_vector_type(8)))  _Float16  v8h;
typedef __attribute__((ext_vector_type(16))) _Float16  v16h;

#define SHUF16(lo, hi) __builtin_shufflevector((lo), (hi), 0,1,2,3,4,5,6,7,8,9,10,11,12,13,14,15)
#define SHUF8(lo, hi)  __builtin_shufflevector((lo), (hi), 0,1,2,3,4,5,6,7)

__device__ __forceinline__ v16h load_a16(const _Float16* p) {
    v8h lo = *(const v8h*)(p);
    v8h hi = *(const v8h*)(p + 8);
    return SHUF16(lo, hi);
}
__device__ __forceinline__ v16h load_b16(const _Float16* p) {
    v8h lo = *(const v8h*)(p);
    v8h hi = *(const v8h*)(p + 8);
    return SHUF16(lo, hi);
}
__device__ __forceinline__ v8i load_a8(const i8* p) {
    v4i lo = *(const v4i*)(p);
    v4i hi = *(const v4i*)(p + 32);
    return SHUF8(lo, hi);
}
__device__ __forceinline__ v8i load_b8(const i8* p) {
    v4i lo = *(const v4i*)(p);
    v4i hi = *(const v4i*)(p + 32);
    return SHUF8(lo, hi);
}

__global__ void __launch_bounds__(256) quant_rows_kernel(
    const void* __restrict__ src, int srcHalf,
    i8* __restrict__ dst, float* __restrict__ scales, int C)
{
    const int row = blockIdx.x;
    const int tid = threadIdx.x;
    __shared__ float red[256];

    float m = 0.0f;
    for (int c = tid; c < C; c += 256) {
        float v = srcHalf ? (float)((const _Float16*)src)[(size_t)row * C + c]
                          : ((const float*)src)[(size_t)row * C + c];
        m = fmaxf(m, fabsf(v));
    }
    red[tid] = m;
    __syncthreads();
    for (int s = 128; s > 0; s >>= 1) {
        if (tid < s) red[tid] = fmaxf(red[tid], red[tid + s]);
        __syncthreads();
    }
    const float scale = fmaxf(red[0] * (1.0f / 127.0f), 1e-8f);
    if (tid == 0) scales[row] = scale;
    const float inv = 1.0f / scale;
    for (int c = tid; c < C; c += 256) {
        float v = srcHalf ? (float)((const _Float16*)src)[(size_t)row * C + c]
                          : ((const float*)src)[(size_t)row * C + c];
        float q = rintf(v * inv);
        q = fminf(127.0f, fmaxf(-127.0f, q));
        dst[(size_t)row * C + c] = (i8)q;
    }
}

__global__ void __launch_bounds__(256) cvt_f32_to_f16_kernel(
    const float* __restrict__ src, _Float16* __restrict__ dst, int n)
{
    int i = blockIdx.x * 256 + threadIdx.x;
    if (i < n) dst[i] = (_Float16)src[i];
}

__global__ void __launch_bounds__(32) gemm_w8a8_kernel(
    const i8* __restrict__ qa, const float* __restrict__ sa,
    const i8* __restrict__ qb, const float* __restrict__ sb,
    const float* __restrict__ bias, void* __restrict__ out,
    int mode, int C, int Nper, int H, int D, int Ocols)
{
    const int lane = threadIdx.x;
    const int col  = lane & 15;
    const int hi   = lane >> 4;
    const int r0   = blockIdx.x * 32;
    const int o0   = blockIdx.y * 64;

    const i8* aR[2];
    aR[0] = qa + (size_t)(r0 + col) * C + hi * 16;
    aR[1] = qa + (size_t)(r0 + 16 + col) * C + hi * 16;
    const i8* bCol[4];
#pragma unroll
    for (int t = 0; t < 4; ++t)
        bCol[t] = qb + (size_t)(o0 + t * 16 + col) * C + hi * 16;

    v8i Ab[2][2], Bb[2][4];
#pragma unroll
    for (int g = 0; g < 2; ++g) Ab[0][g] = load_a8(aR[g]);
#pragma unroll
    for (int t = 0; t < 4; ++t) Bb[0][t] = load_b8(bCol[t]);

    v8i acc[2][4] = {};
#pragma unroll 2
    for (int k0 = 0, p = 0; k0 < C; k0 += 64, p ^= 1) {
        const int kn = k0 + 64;
#pragma unroll
        for (int g = 0; g < 2; ++g) Ab[p ^ 1][g] = load_a8(aR[g] + kn);
#pragma unroll
        for (int t = 0; t < 4; ++t) Bb[p ^ 1][t] = load_b8(bCol[t] + kn);
#pragma unroll
        for (int t = 0; t < 4; ++t)
#pragma unroll
            for (int g = 0; g < 2; ++g)
                acc[g][t] = __builtin_amdgcn_wmma_i32_16x16x64_iu8(
                    true, Ab[p][g], true, Bb[p][t], acc[g][t], false, false);
    }

    const int bb = r0 / Nper;
    const int nb = r0 % Nper;
#pragma unroll
    for (int g = 0; g < 2; ++g) {
        float srow[8];
#pragma unroll
        for (int r = 0; r < 8; ++r) srow[r] = sa[r0 + g * 16 + r + hi * 8];
#pragma unroll
        for (int t = 0; t < 4; ++t) {
            const int o    = o0 + t * 16 + col;
            const float sw = sb[o];
            const float bc = bias[o];
            const int hh = o / D, d = o % D;
#pragma unroll
            for (int r = 0; r < 8; ++r) {
                float y = (float)acc[g][t][r] * (srow[r] * sw) + bc;
                const int rr = g * 16 + r + hi * 8;
                if (mode == 0) {
                    ((_Float16*)out)[(((size_t)bb * H + hh) * Nper + (nb + rr)) * D + d] =
                        (_Float16)y;
                } else {
                    ((float*)out)[(size_t)(r0 + rr) * Ocols + o] = y;
                }
            }
        }
    }
}

__global__ void __launch_bounds__(32) gemm_f16_kernel(
    const _Float16* __restrict__ a, const _Float16* __restrict__ w,
    const float* __restrict__ bias, _Float16* __restrict__ out,
    int mode, int C, int L, int H, int D)
{
    const int lane = threadIdx.x;
    const int col  = lane & 15;
    const int hi   = lane >> 4;
    const int r0   = blockIdx.x * 32;
    const int o0   = blockIdx.y * 64;

    const _Float16* aR[2];
    aR[0] = a + (size_t)(r0 + col) * C + hi * 16;
    aR[1] = a + (size_t)(r0 + 16 + col) * C + hi * 16;
    const _Float16* bCol[4];
#pragma unroll
    for (int t = 0; t < 4; ++t)
        bCol[t] = w + (size_t)(o0 + t * 16 + col) * C + hi * 16;

    v16h Ab[2][2], Bb[2][4];
#pragma unroll
    for (int g = 0; g < 2; ++g) Ab[0][g] = load_a16(aR[g]);
#pragma unroll
    for (int t = 0; t < 4; ++t) Bb[0][t] = load_b16(bCol[t]);

    v8f acc[2][4] = {};
#pragma unroll 2
    for (int k0 = 0, p = 0; k0 < C; k0 += 32, p ^= 1) {
        const int kn = k0 + 32;
#pragma unroll
        for (int g = 0; g < 2; ++g) Ab[p ^ 1][g] = load_a16(aR[g] + kn);
#pragma unroll
        for (int t = 0; t < 4; ++t) Bb[p ^ 1][t] = load_b16(bCol[t] + kn);
#pragma unroll
        for (int t = 0; t < 4; ++t)
#pragma unroll
            for (int g = 0; g < 2; ++g)
                acc[g][t] = __builtin_amdgcn_wmma_f32_16x16x32_f16(
                    false, Ab[p][g], false, Bb[p][t], (short)0, acc[g][t],
                    false, false);
    }

    const int bb = r0 / L;
    const int lb = r0 % L;
#pragma unroll
    for (int t = 0; t < 4; ++t) {
        const int o    = o0 + t * 16 + col;
        const float bc = bias[o];
        const int hh = o / D, d = o % D;
#pragma unroll
        for (int g = 0; g < 2; ++g)
#pragma unroll
            for (int r = 0; r < 8; ++r) {
                const int l = lb + g * 16 + r + hi * 8;
                float y = acc[g][t][r] + bc;
                if (mode == 0)
                    out[(((size_t)bb * H + hh) * L + l) * D + d] = (_Float16)y;
                else
                    out[(((size_t)bb * H + hh) * D + d) * L + l] = (_Float16)y;
            }
    }
}

__global__ void __launch_bounds__(32) attn_kernel(
    const _Float16* __restrict__ q, const _Float16* __restrict__ k,
    const _Float16* __restrict__ vT, const float* __restrict__ mask,
    _Float16* __restrict__ o16, int B, int H, int N, int L, int D, float scale)
{
    __shared__ float S[16 * 512];
    _Float16* Sh = (_Float16*)S;

    const int lane = threadIdx.x;
    const int col  = lane & 15;
    const int hi   = lane >> 4;
    const int n0   = blockIdx.x * 16;
    const int h    = blockIdx.y;
    const int b    = blockIdx.z;
    const size_t bh = (size_t)b * H + h;

    const _Float16* qrow = q + (bh * N + n0 + col) * D + hi * 16;
    v16h Af[4];
#pragma unroll
    for (int kk = 0; kk < 4; ++kk) Af[kk] = load_a16(qrow + kk * 32);

    const _Float16* kbase = k + bh * L * D + (size_t)col * D + hi * 16;
    v16h Kb[2][4];
#pragma unroll
    for (int kk = 0; kk < 4; ++kk) Kb[0][kk] = load_b16(kbase + kk * 32);
#pragma unroll 2
    for (int lt = 0, p = 0; lt < 32; ++lt, p ^= 1) {
        const _Float16* knxt = kbase + (size_t)(lt + 1) * 16 * D;
#pragma unroll
        for (int kk = 0; kk < 4; ++kk) Kb[p ^ 1][kk] = load_b16(knxt + kk * 32);
        v8f acc = {};
#pragma unroll
        for (int kk = 0; kk < 4; ++kk)
            acc = __builtin_amdgcn_wmma_f32_16x16x32_f16(
                false, Af[kk], false, Kb[p][kk], (short)0, acc, false, false);
#pragma unroll
        for (int r = 0; r < 8; ++r)
            S[(r + hi * 8) * 512 + lt * 16 + col] = acc[r];
    }
    __syncthreads();

    const float* mrow = mask + ((size_t)b * N + n0) * L;
    for (int r = 0; r < 16; ++r) {
        float vals[16];
        float m = -1e30f;
#pragma unroll
        for (int j = 0; j < 16; ++j) {
            const int c = lane + j * 32;
            float vv = S[r * 512 + c] * scale + mrow[(size_t)r * L + c];
            vals[j] = vv;
            m = fmaxf(m, vv);
        }
        for (int off = 16; off > 0; off >>= 1)
            m = fmaxf(m, __shfl_xor(m, off, 32));
        float sum = 0.0f;
#pragma unroll
        for (int j = 0; j < 16; ++j) {
            float e = __expf(vals[j] - m);
            vals[j] = e;
            sum += e;
        }
        for (int off = 16; off > 0; off >>= 1)
            sum += __shfl_xor(sum, off, 32);
        const float inv = 1.0f / sum;
#pragma unroll
        for (int j = 0; j < 16; ++j)
            Sh[r * 512 + lane + j * 32] = (_Float16)(vals[j] * inv);
    }
    __syncthreads();

    v8f oacc[8] = {};
    const _Float16* vbase = vT + bh * D * L + (size_t)col * L + hi * 16;
    v16h Vb[8];
#pragma unroll
    for (int dt = 0; dt < 4; ++dt) Vb[dt] = load_b16(vbase + (size_t)dt * 16 * L);
    for (int kt = 0; kt < 16; ++kt) {
        const int k0 = kt * 32;
        const _Float16* pp = Sh + (size_t)col * 512 + k0 + hi * 16;
        v16h Ap = load_a16(pp);
#pragma unroll
        for (int dt = 4; dt < 8; ++dt)
            Vb[dt] = load_b16(vbase + (size_t)dt * 16 * L + k0);
#pragma unroll
        for (int dt = 0; dt < 4; ++dt)
            oacc[dt] = __builtin_amdgcn_wmma_f32_16x16x32_f16(
                false, Ap, false, Vb[dt], (short)0, oacc[dt], false, false);
#pragma unroll
        for (int dt = 0; dt < 4; ++dt)
            Vb[dt] = load_b16(vbase + (size_t)dt * 16 * L + k0 + 32);
#pragma unroll
        for (int dt = 4; dt < 8; ++dt)
            oacc[dt] = __builtin_amdgcn_wmma_f32_16x16x32_f16(
                false, Ap, false, Vb[dt], (short)0, oacc[dt], false, false);
    }

    const int C = H * D;
    _Float16* obase = o16 + ((size_t)b * N + n0) * C + h * D;
#pragma unroll
    for (int dt = 0; dt < 8; ++dt)
#pragma unroll
        for (int r = 0; r < 8; ++r)
            obase[(size_t)(r + hi * 8) * C + dt * 16 + col] = (_Float16)oacc[dt][r];
}

extern "C" void kernel_launch(void* const* d_in, const int* in_sizes, int n_in,
                              void* d_out, int out_size, void* d_ws, size_t ws_size,
                              hipStream_t stream) {
    (void)in_sizes; (void)n_in; (void)out_size; (void)ws_size;

    constexpr int B = 2, N = 4096, L = 512, C = 2048, H = 16, D = 128;

    const float* x    = (const float*)d_in[0];
    const float* cond = (const float*)d_in[1];
    const float* mask = (const float*)d_in[2];
    const float* wq   = (const float*)d_in[3];
    const float* bq   = (const float*)d_in[4];
    const float* wk   = (const float*)d_in[5];
    const float* bk   = (const float*)d_in[6];
    const float* wv   = (const float*)d_in[7];
    const float* bv   = (const float*)d_in[8];
    const float* wo   = (const float*)d_in[9];
    const float* bo   = (const float*)d_in[10];
    float* out = (float*)d_out;

    char* ws = (char*)d_ws;
    size_t off = 0;
    auto alloc = [&](size_t bytes) -> char* {
        char* p = ws + off;
        off = (off + bytes + 255) & ~(size_t)255;
        return p;
    };

    i8*       qact = (i8*)alloc((size_t)B * N * C);
    float*    sact = (float*)alloc((size_t)B * N * 4);
    i8*       qwq  = (i8*)alloc((size_t)C * C);
    float*    swq  = (float*)alloc((size_t)C * 4);
    i8*       qwo  = (i8*)alloc((size_t)C * C);
    float*    swo  = (float*)alloc((size_t)C * 4);
    _Float16* q16  = (_Float16*)alloc((size_t)B * N * C * 2);
    _Float16* c16  = (_Float16*)alloc((size_t)B * L * C * 2);
    _Float16* wk16 = (_Float16*)alloc((size_t)C * C * 2);
    _Float16* wv16 = (_Float16*)alloc((size_t)C * C * 2);
    _Float16* k16  = (_Float16*)alloc((size_t)B * H * L * D * 2);
    _Float16* vT16 = (_Float16*)alloc((size_t)B * H * D * L * 2);
    _Float16* o16  = (_Float16*)alloc((size_t)B * N * C * 2);
    (void)alloc(4096);

    quant_rows_kernel<<<B * N, 256, 0, stream>>>(x, 0, qact, sact, C);
    quant_rows_kernel<<<C, 256, 0, stream>>>(wq, 0, qwq, swq, C);
    quant_rows_kernel<<<C, 256, 0, stream>>>(wo, 0, qwo, swo, C);

    int nc = B * L * C, nw = C * C;
    cvt_f32_to_f16_kernel<<<(nc + 255) / 256, 256, 0, stream>>>(cond, c16, nc);
    cvt_f32_to_f16_kernel<<<(nw + 255) / 256, 256, 0, stream>>>(wk, wk16, nw);
    cvt_f32_to_f16_kernel<<<(nw + 255) / 256, 256, 0, stream>>>(wv, wv16, nw);

    gemm_w8a8_kernel<<<dim3(B * N / 32, C / 64), 32, 0, stream>>>(
        qact, sact, qwq, swq, bq, q16, 0, C, N, H, D, C);

    gemm_f16_kernel<<<dim3(B * L / 32, C / 64), 32, 0, stream>>>(
        c16, wk16, bk, k16, 0, C, L, H, D);
    gemm_f16_kernel<<<dim3(B * L / 32, C / 64), 32, 0, stream>>>(
        c16, wv16, bv, vT16, 1, C, L, H, D);

    attn_kernel<<<dim3(N / 16, H, B), 32, 0, stream>>>(
        q16, k16, vT16, mask, o16, B, H, N, L, D, 0.08838834764831845f);

    quant_rows_kernel<<<B * N, 256, 0, stream>>>(o16, 1, qact, sact, C);
    gemm_w8a8_kernel<<<dim3(B * N / 32, C / 64), 32, 0, stream>>>(
        qact, sact, qwo, swo, bo, out, 1, C, N, H, D, C);
}
